// Attention_47124381171831
// MI455X (gfx1250) — matrix operand order repaired and checked by reading, not compiled, not run on hardware
//
#include <hip/hip_runtime.h>
#include <hip/hip_bf16.h>

// ---------------------------------------------------------------------------
// CDNA5 (gfx1250) fused relative-position attention block, bf16 WMMA path.
// All operands stored in contiguous-k (k = 16h + s) WMMA fragment order -> every fragment load
// is one contiguous 32-byte per-lane read (2x b128), zero repack VALU.
// ---------------------------------------------------------------------------

typedef __attribute__((ext_vector_type(16))) __bf16 v16bf;
typedef __attribute__((ext_vector_type(8)))  float  v8f;

#define B_DIM   16
#define C_DIM   256
#define N_TOK   1024
#define H_DIM   8
#define DK_DIM  32
#define DV_DIM  64
#define VINNER  512
#define EPSV    1e-5f
#define SCALE_F 0.17677669529663689f   // 32^-0.5
#define INVSC_F 5.656854249492381f     // 32^0.5

// float -> bf16, round-to-nearest-even
__device__ __forceinline__ __bf16 f2bf(float f) {
    unsigned u = __builtin_bit_cast(unsigned, f);
    unsigned short s = (unsigned short)((u + 0x7FFFu + ((u >> 16) & 1u)) >> 16);
    return __builtin_bit_cast(__bf16, s);
}

__device__ __forceinline__ v8f wmma_bf16(v16bf a, v16bf b, v8f c) {
    return __builtin_amdgcn_wmma_f32_16x16x32_bf16(
        false, a, false, b, (short)0, c, false, false);
}

// Load 16 contiguous bf16 (32B per lane) -> one fragment register set.
__device__ __forceinline__ v16bf load_frag16(const __bf16* p) {
    v16bf a;
#pragma unroll
    for (int j = 0; j < 16; ++j) a[j] = p[j];
    return a;
}

// Packed A-fragment offset for element (o,k) of an OxK matrix, K in 32-slices.
// 16-bit A 16x32 (contiguous k order): lane = (o&15) + 16*((k&31)>=16),
// half index = k&15.
__device__ __forceinline__ size_t frag_pack_off(int o, int k, int kslices) {
    int ot = o >> 4, ks = k >> 5;
    int lane = (o & 15) + (((k & 31) >= 16) ? 16 : 0);
    int idx  = k & 15;
    return ((size_t)(ot * kslices + ks) * 32 + lane) * 16 + idx;
}

__device__ __forceinline__ float half16_max(float v) {
    v = fmaxf(v, __shfl_xor(v, 1, 32));
    v = fmaxf(v, __shfl_xor(v, 2, 32));
    v = fmaxf(v, __shfl_xor(v, 4, 32));
    v = fmaxf(v, __shfl_xor(v, 8, 32));
    return v;
}
__device__ __forceinline__ float half16_sum(float v) {
    v += __shfl_xor(v, 1, 32);
    v += __shfl_xor(v, 2, 32);
    v += __shfl_xor(v, 4, 32);
    v += __shfl_xor(v, 8, 32);
    return v;
}

// ---------------------------------------------------------------------------
// Prep kernels: fold BN, pack weights into A-fragment order, transpose x.
// ---------------------------------------------------------------------------
__global__ void prep_qkv_w(const float* __restrict__ Wq, const float* __restrict__ Wk,
                           const float* __restrict__ Wv,
                           const float* qg, const float* qb, const float* qm, const float* qv,
                           const float* kg, const float* kbp, const float* km, const float* kv,
                           const float* vg, const float* vbp, const float* vm, const float* vv,
                           __bf16* __restrict__ Wpk, float* __restrict__ tvec) {
    int i = blockIdx.x * blockDim.x + threadIdx.x;
    if (i >= 1024 * 256) return;
    int o = i >> 8, c = i & 255;
    float w, g, bb, m, var;
    if (o < 256)      { w = Wq[o * 256 + c];          g = qg[o];  bb = qb[o];   m = qm[o];  var = qv[o]; }
    else if (o < 512) { int oo = o - 256; w = Wk[oo * 256 + c]; g = kg[oo]; bb = kbp[oo]; m = km[oo]; var = kv[oo]; }
    else              { int oo = o - 512; w = Wv[oo * 256 + c]; g = vg[oo]; bb = vbp[oo]; m = vm[oo]; var = vv[oo]; }
    float s = g * rsqrtf(var + EPSV);
    Wpk[frag_pack_off(o, c, 8)] = f2bf(w * s);
    if (c == 0) tvec[o] = bb - m * s;
}

__global__ void prep_out_w(const float* __restrict__ Wo, const float* __restrict__ bo,
                           const float* og, const float* ob, const float* om, const float* ov,
                           __bf16* __restrict__ Wopk, float* __restrict__ tob) {
    int i = blockIdx.x * blockDim.x + threadIdx.x;
    if (i >= 256 * 512) return;
    int o = i >> 9, c = i & 511;
    float s = og[o] * rsqrtf(ov[o] + EPSV);
    Wopk[frag_pack_off(o, c, 16)] = f2bf(Wo[i] * s);
    if (c == 0) tob[o] = (bo[o] - om[o]) * s + ob[o];
}

// x (b,c,n) f32 -> xbt (b,n,c) bf16 (k-contiguous for B fragments)
__global__ void conv_x_bf16(const float* __restrict__ x, __bf16* __restrict__ xbt) {
    int i = blockIdx.x * blockDim.x + threadIdx.x;
    if (i >= B_DIM * C_DIM * N_TOK) return;
    int c = i & 255;
    int n = (i >> 8) & 1023;
    int b = i >> 18;
    xbt[i] = f2bf(x[((size_t)b * 256 + c) * 1024 + n]);
}

// ---------------------------------------------------------------------------
// QKV projection. o0 is block-uniform -> scalar section branch in epilogue.
//   q -> qpk   : packed A-fragment order per (b,h,q-tile)  [attention A side]
//   k -> kbuf  : (b,h,key,d) d-contiguous                  [attention B side]
//   v -> vbuf  : (b,ch,key)  key-contiguous                [attention B side]
// ---------------------------------------------------------------------------
__global__ __launch_bounds__(256) void qkv_gemm(const __bf16* __restrict__ Wpk,
                                                const float* __restrict__ tvec,
                                                const __bf16* __restrict__ xbt,
                                                __bf16* __restrict__ qpk,
                                                __bf16* __restrict__ kbuf,
                                                __bf16* __restrict__ vbuf) {
    int lane = threadIdx.x & 31;
    int wv   = threadIdx.x >> 5;
    int b    = blockIdx.x >> 7;          // 2048 blocks: b(4) | ot(6) | nh(1)
    int ot   = (blockIdx.x >> 1) & 63;
    int nh   = blockIdx.x & 1;
    int o0   = ot * 16;                  // uniform (SGPR)
    int n0   = (nh * 8 + wv) * 64;
    int col = lane & 15, halfid = lane >> 4, koff = halfid * 16;

    v8f acc[4] = {};
    for (int ks = 0; ks < 8; ++ks) {
        v16bf a = load_frag16(Wpk + ((size_t)(ot * 8 + ks) * 32 + lane) * 16);
#pragma unroll
        for (int sub = 0; sub < 4; ++sub) {
            v16bf bf = load_frag16(
                xbt + ((size_t)b * 1024 + n0 + sub * 16 + col) * 256 + ks * 32 + koff);
            acc[sub] = wmma_bf16(a, bf, acc[sub]);
        }
    }

    float tv[8];
#pragma unroll
    for (int r = 0; r < 8; ++r) tv[r] = tvec[o0 + r + 8 * halfid];

    if (o0 < 256) {                       // Q: packed A-fragment order
        int h   = o0 >> 5;
        int grp = (o0 >> 4) & 1;
        size_t tb = (size_t)(b * 8 + h) * 64;
#pragma unroll
        for (int sub = 0; sub < 4; ++sub) {
            size_t fb = ((tb + (n0 >> 4) + sub) * 32 + col + 16 * grp) * 16 + 8 * halfid;
#pragma unroll
            for (int r = 0; r < 8; ++r)
                qpk[fb + r] = f2bf(acc[sub][r] + tv[r]);
        }
    } else if (o0 < 512) {                // K: (b,h,key,d)
        int oo0 = o0 - 256;
        int h   = oo0 >> 5;
        int d0  = (oo0 & 31) + 8 * halfid;
#pragma unroll
        for (int sub = 0; sub < 4; ++sub) {
            int n = n0 + sub * 16 + col;
            size_t fb = ((size_t)(b * 8 + h) * 1024 + n) * 32 + d0;
#pragma unroll
            for (int r = 0; r < 8; ++r)
                kbuf[fb + r] = f2bf(acc[sub][r] + tv[r]);
        }
    } else {                              // V: (b,ch,key)
        int ov0 = (o0 - 512) + 8 * halfid;
#pragma unroll
        for (int sub = 0; sub < 4; ++sub) {
            int n = n0 + sub * 16 + col;
#pragma unroll
            for (int r = 0; r < 8; ++r)
                vbuf[((size_t)b * 512 + ov0 + r) * 1024 + n] = f2bf(acc[sub][r] + tv[r]);
        }
    }
}

// ---------------------------------------------------------------------------
// Flash-style attention: one wave per (b,h,16-query tile). Scores stay in
// VGPRs; P tile round-trips 1KB LDS in packed fragment order.
// ---------------------------------------------------------------------------
__global__ __launch_bounds__(256) void attn_kernel(const __bf16* __restrict__ qpk,
                                                   const __bf16* __restrict__ kbuf,
                                                   const __bf16* __restrict__ vbuf,
                                                   const float* __restrict__ pos_tab,
                                                   const int* __restrict__ pos_idx,
                                                   __bf16* __restrict__ attbt) {
    __shared__ __bf16 plds[8 * 512];      // 8 waves x packed 16x32 P tile

    int lane = threadIdx.x & 31;
    int wv   = threadIdx.x >> 5;
    int t    = blockIdx.x * 8 + wv;       // 8192 waves
    int b    = t >> 9;
    int rem  = t & 511;
    int h    = rem >> 6;
    int qt   = rem & 63;
    int q0   = qt * 16;
    int bh   = b * 8 + h;
    int col = lane & 15, halfid = lane >> 4, koff = halfid * 16;
    __bf16* myP = plds + wv * 512;

    v16bf qa = load_frag16(qpk + (((size_t)bh * 64 + qt) * 32 + lane) * 16);

    const __bf16* kbase = kbuf + (size_t)bh * 1024 * 32;
    const __bf16* vbase = vbuf + ((size_t)b * 512 + h * 64) * 1024;

    float mrow[8], lrow[8];
#pragma unroll
    for (int r = 0; r < 8; ++r) { mrow[r] = -1e30f; lrow[r] = 0.f; }
    v8f oacc[4] = {};

    int mlo = (col >= 8) ? 16 : 0;        // packed-P lane group
    int ilo = col & 7;

    for (int j0 = 0; j0 < 1024; j0 += 32) {
        // ---- scores: two 16x16 WMMA tiles ----------------------------------
        v8f z = {};
        v16bf kf0 = load_frag16(kbase + (size_t)(j0 + col) * 32 + koff);
        v8f sc0 = wmma_bf16(qa, kf0, z);
        v16bf kf1 = load_frag16(kbase + (size_t)(j0 + 16 + col) * 32 + koff);
        v8f sc1 = wmma_bf16(qa, kf1, z);

        if (j0 + 32 < 1024) {             // global_prefetch_b8 next K tile
            __builtin_prefetch(kbase + (size_t)(j0 + 32 + col) * 32, 0, 1);
            __builtin_prefetch(vbase + (size_t)col * 1024 + j0 + 32, 0, 1);
        }

        float s0[8], s1[8];
#pragma unroll
        for (int r = 0; r < 8; ++r) {
            int qi = q0 + r + 8 * halfid;
            float b0 = pos_tab[pos_idx[qi * 1024 + j0 + col] * 8 + h];
            float b1 = pos_tab[pos_idx[qi * 1024 + j0 + 16 + col] * 8 + h];
            s0[r] = sc0[r] * SCALE_F + b0 * INVSC_F;
            s1[r] = sc1[r] * SCALE_F + b1 * INVSC_F;
        }

        // ---- online softmax, P tile written in packed A-frag order ---------
        float crow[8];
#pragma unroll
        for (int r = 0; r < 8; ++r) {
            float mx = half16_max(fmaxf(s0[r], s1[r]));
            float mn = fmaxf(mrow[r], mx);
            float corr = __expf(mrow[r] - mn);
            float p0 = __expf(s0[r] - mn);
            float p1 = __expf(s1[r] - mn);
            lrow[r] = lrow[r] * corr + half16_sum(p0 + p1);
            mrow[r] = mn;
            crow[r] = corr;
            int mm = r + 8 * halfid;
            myP[mm * 16 + col]        = f2bf(p0);   // k-group 0
            myP[(mm + 16) * 16 + col] = f2bf(p1);   // k-group 1
        }
#pragma unroll
        for (int sub = 0; sub < 4; ++sub)
#pragma unroll
            for (int r = 0; r < 8; ++r) oacc[sub][r] *= crow[r];

        // CDNA5 split counters: LDS stores must land before packed reload
        asm volatile("s_wait_dscnt 0" ::: "memory");

        // ---- P(16x32) @ V(32x64): 4 WMMAs ----------------------------------
        v16bf pa = load_frag16(myP + lane * 16);
#pragma unroll
        for (int sub = 0; sub < 4; ++sub) {
            v16bf vf = load_frag16(vbase + (size_t)(sub * 16 + col) * 1024 + j0 + koff);
            oacc[sub] = wmma_bf16(pa, vf, oacc[sub]);
        }
    }

    // ---- normalize, ReLU, store transposed (b,q,ch) for out_gemm B frags ---
#pragma unroll
    for (int sub = 0; sub < 4; ++sub) {
        int ch = h * 64 + sub * 16 + col;
#pragma unroll
        for (int r = 0; r < 8; ++r) {
            int q = q0 + r + 8 * halfid;
            float val = oacc[sub][r] / lrow[r];
            attbt[((size_t)b * 1024 + q) * 512 + ch] = f2bf(fmaxf(val, 0.f));
        }
    }
}

// ---------------------------------------------------------------------------
// Output projection: [256 x 1024] = Wo'[256x512] @ relu(att)[512x1024] / batch
// ---------------------------------------------------------------------------
__global__ __launch_bounds__(256) void out_gemm(const __bf16* __restrict__ Wopk,
                                                const float* __restrict__ tob,
                                                const __bf16* __restrict__ attbt,
                                                float* __restrict__ out) {
    int lane = threadIdx.x & 31;
    int wv   = threadIdx.x >> 5;
    int t    = blockIdx.x * 8 + wv;       // 4096 waves
    int b    = t >> 8;
    int rem  = t & 255;
    int ot   = rem >> 4;
    int n0   = (rem & 15) * 64;
    int o0   = ot * 16;
    int col = lane & 15, halfid = lane >> 4, koff = halfid * 16;

    v8f acc[4] = {};
    for (int ks = 0; ks < 16; ++ks) {
        v16bf a = load_frag16(Wopk + ((size_t)(ot * 16 + ks) * 32 + lane) * 16);
#pragma unroll
        for (int sub = 0; sub < 4; ++sub) {
            v16bf bf = load_frag16(
                attbt + ((size_t)b * 1024 + n0 + sub * 16 + col) * 512 + ks * 32 + koff);
            acc[sub] = wmma_bf16(a, bf, acc[sub]);
        }
    }

#pragma unroll
    for (int sub = 0; sub < 4; ++sub) {
#pragma unroll
        for (int r = 0; r < 8; ++r) {
            int o = o0 + r + 8 * halfid;
            int n = n0 + sub * 16 + col;
            out[((size_t)b * 256 + o) * 1024 + n] = acc[sub][r] + tob[o];
        }
    }
}

// ---------------------------------------------------------------------------
extern "C" void kernel_launch(void* const* d_in, const int* in_sizes, int n_in,
                              void* d_out, int out_size, void* d_ws, size_t ws_size,
                              hipStream_t stream) {
    const float* x       = (const float*)d_in[0];
    const float* Wq      = (const float*)d_in[1];
    const float* Wk      = (const float*)d_in[2];
    const float* Wv      = (const float*)d_in[3];
    const float* Wo      = (const float*)d_in[4];
    const float* bo      = (const float*)d_in[5];
    const float* pos_tab = (const float*)d_in[6];
    const float* q_g = (const float*)d_in[7],  *q_b = (const float*)d_in[8];
    const float* q_m = (const float*)d_in[9],  *q_v = (const float*)d_in[10];
    const float* k_g = (const float*)d_in[11], *k_b = (const float*)d_in[12];
    const float* k_m = (const float*)d_in[13], *k_v = (const float*)d_in[14];
    const float* v_g = (const float*)d_in[15], *v_b = (const float*)d_in[16];
    const float* v_m = (const float*)d_in[17], *v_v = (const float*)d_in[18];
    const float* o_g = (const float*)d_in[19], *o_b = (const float*)d_in[20];
    const float* o_m = (const float*)d_in[21], *o_v = (const float*)d_in[22];
    const int*   pos_idx = (const int*)d_in[23];
    float* out = (float*)d_out;

    char* ws = (char*)d_ws;
    size_t off = 0;
    auto alloc = [&](size_t bytes) -> void* {
        off = (off + 255) & ~(size_t)255;
        void* p = ws + off;
        off += bytes;
        return p;
    };

    __bf16* xbt  = (__bf16*)alloc((size_t)B_DIM * C_DIM * N_TOK * 2);           // 8 MB
    __bf16* Wpk  = (__bf16*)alloc((size_t)1024 * 256 * 2);                      // 512 KB
    float*  tvec = (float*) alloc(1024 * 4);
    __bf16* Wopk = (__bf16*)alloc((size_t)256 * 512 * 2);                       // 256 KB
    float*  tob  = (float*) alloc(256 * 4);
    __bf16* qpk  = (__bf16*)alloc((size_t)B_DIM * H_DIM * N_TOK * DK_DIM * 2);  // 8 MB
    __bf16* kbuf = (__bf16*)alloc((size_t)B_DIM * H_DIM * N_TOK * DK_DIM * 2);  // 8 MB
    __bf16* vbuf = (__bf16*)alloc((size_t)B_DIM * VINNER * N_TOK * 2);          // 16 MB
    __bf16* attbt= (__bf16*)alloc((size_t)B_DIM * VINNER * N_TOK * 2);          // 16 MB

    conv_x_bf16<<<(B_DIM * C_DIM * N_TOK + 255) / 256, 256, 0, stream>>>(x, xbt);
    prep_qkv_w<<<(1024 * 256 + 255) / 256, 256, 0, stream>>>(
        Wq, Wk, Wv, q_g, q_b, q_m, q_v, k_g, k_b, k_m, k_v, v_g, v_b, v_m, v_v, Wpk, tvec);
    prep_out_w<<<(256 * 512 + 255) / 256, 256, 0, stream>>>(
        Wo, bo, o_g, o_b, o_m, o_v, Wopk, tob);

    qkv_gemm<<<2048, 256, 0, stream>>>(Wpk, tvec, xbt, qpk, kbuf, vbuf);
    attn_kernel<<<1024, 256, 0, stream>>>(qpk, kbuf, vbuf, pos_tab, pos_idx, attbt);
    out_gemm<<<512, 256, 0, stream>>>(Wopk, tob, attbt, out);
}
